// GINModel_32049045963189
// MI455X (gfx1250) — matrix operand order repaired and checked by reading, not compiled, not run on hardware
//
#include <hip/hip_runtime.h>

#define DHID 128

typedef __attribute__((ext_vector_type(16))) __bf16 v16bf;
typedef __attribute__((ext_vector_type(8)))  __bf16 v8bf;
typedef __attribute__((ext_vector_type(8)))  float  v8f;

// ---------------------------------------------------------------------------
// Weight prep: f32 W1/W2 [L][128][128] -> bf16, swizzled into WMMA B-operand
// blocks. Block (l, mat, kb, nt) holds 32 lanes x 16 bf16:
//   lane L -> column n = nt*16 + (L%16), element i -> k = kb*32 + 16*(i/8) + 8*(L/16) + (i%8)
// Each lane's 16 values are contiguous (32B) => a clean pair of b128 loads.
// ---------------------------------------------------------------------------
__global__ void gin_wprep(const float* __restrict__ W1,
                          const float* __restrict__ W2,
                          __bf16* __restrict__ wsw, int total)
{
    int idx = blockIdx.x * blockDim.x + threadIdx.x;
    if (idx >= total) return;
    int i    = idx & 15;
    int lane = (idx >> 4) & 31;
    int nt   = (idx >> 9) & 7;
    int kb   = (idx >> 12) & 3;
    int mat  = (idx >> 14) & 1;
    int l    = idx >> 15;
    int k = kb * 32 + 16 * (i >> 3) + 8 * (lane >> 4) + (i & 7);
    int n = nt * 16 + (lane & 15);
    const float* W = mat ? W2 : W1;
    wsw[idx] = (__bf16)W[((size_t)l * DHID + k) * DHID + n];
}

// ---------------------------------------------------------------------------
// Edge scatter: one wave per edge. Edge index forced wave-uniform with
// readfirstlane so src/dst become scalar loads and the 512B row gather uses
// the SADDR addressing form. Lane l adds x[src][4l..4l+3] into agg[dst] with
// 4 global_atomic_add_f32 (agg is L2-resident: 25.6MB << 192MB L2).
// ---------------------------------------------------------------------------
__global__ void gin_scatter(const float* __restrict__ x,
                            const int* __restrict__ src,
                            const int* __restrict__ dst,
                            float* __restrict__ agg, int nedges)
{
    int e    = (blockIdx.x * blockDim.x + threadIdx.x) >> 5;
    int lane = threadIdx.x & 31;
    e = __builtin_amdgcn_readfirstlane(e);     // wave-uniform edge id
    if (e >= nedges) return;
    long s = src[e];
    long d = dst[e];
    const float4 v = *(const float4*)(x + s * DHID + lane * 4);
    float* p = agg + d * DHID + lane * 4;
    __hip_atomic_fetch_add(p + 0, v.x, __ATOMIC_RELAXED, __HIP_MEMORY_SCOPE_AGENT);
    __hip_atomic_fetch_add(p + 1, v.y, __ATOMIC_RELAXED, __HIP_MEMORY_SCOPE_AGENT);
    __hip_atomic_fetch_add(p + 2, v.z, __ATOMIC_RELAXED, __HIP_MEMORY_SCOPE_AGENT);
    __hip_atomic_fetch_add(p + 3, v.w, __ATOMIC_RELAXED, __HIP_MEMORY_SCOPE_AGENT);
}

// ---------------------------------------------------------------------------
// Fused MLP: out = relu(h*W1 + b1)*W2 + b2 via v_wmma_f32_16x16x32_bf16.
// 8 waves/block, 16 rows per wave. kb-outer / nt-inner loop order keeps 8
// independent accumulator chains live so B-tile loads overlap WMMA issue.
// Bias is folded into the accumulator init (D = A*B + C).
// ---------------------------------------------------------------------------
__global__ void __launch_bounds__(256)
gin_mlp(const float* __restrict__ hin,
        const __bf16* __restrict__ w1s,
        const float* __restrict__ b1,
        const __bf16* __restrict__ w2s,
        const float* __restrict__ b2,
        float* __restrict__ out, int nrows)
{
    __shared__ __bf16 lds_h[8 * 16 * DHID];   // 8 waves * 16x128 bf16 = 32 KB

    const int wave = threadIdx.x >> 5;
    const int lane = threadIdx.x & 31;
    const int m    = lane & 15;      // row within tile (A) / column (B,C,D)
    const int hi   = lane >> 4;      // half-wave selector

    const long rowBase = (long)blockIdx.x * 128 + wave * 16;
    const long row  = rowBase + m;
    const long srow = (row < nrows) ? row : 0;   // clamp loads; stores guarded

    // ---- Build A for GEMM1 from global f32 (ISA 16-bit A layout:
    //      element i -> K = kb*32 + 16*(i>>3) + 8*hi + (i&7)).
    v16bf a1[4];
    const float* hrow = hin + srow * DHID;
#pragma unroll
    for (int kb = 0; kb < 4; ++kb) {
        const float4 f0 = *(const float4*)(hrow + kb * 32 + 8 * hi);
        const float4 f1 = *(const float4*)(hrow + kb * 32 + 8 * hi + 4);
        const float4 f2 = *(const float4*)(hrow + kb * 32 + 16 + 8 * hi);
        const float4 f3 = *(const float4*)(hrow + kb * 32 + 16 + 8 * hi + 4);
        v16bf a;
        a[0]  = (__bf16)f0.x; a[1]  = (__bf16)f0.y; a[2]  = (__bf16)f0.z; a[3]  = (__bf16)f0.w;
        a[4]  = (__bf16)f1.x; a[5]  = (__bf16)f1.y; a[6]  = (__bf16)f1.z; a[7]  = (__bf16)f1.w;
        a[8]  = (__bf16)f2.x; a[9]  = (__bf16)f2.y; a[10] = (__bf16)f2.z; a[11] = (__bf16)f2.w;
        a[12] = (__bf16)f3.x; a[13] = (__bf16)f3.y; a[14] = (__bf16)f3.z; a[15] = (__bf16)f3.w;
        a1[kb] = a;
    }

    const __bf16* w1p = w1s + (lane << 4);
    const __bf16* w2p = w2s + (lane << 4);

    // ---- GEMM1: accumulators seeded with bias (C column = nt*16 + m).
    v8f acc[8];
#pragma unroll
    for (int nt = 0; nt < 8; ++nt) {
        const float bb = b1[nt * 16 + m];
        acc[nt] = (v8f){bb, bb, bb, bb, bb, bb, bb, bb};
    }
#pragma unroll
    for (int kb = 0; kb < 4; ++kb) {
#pragma unroll
        for (int nt = 0; nt < 8; ++nt) {
            const v16bf b = *(const v16bf*)(w1p + ((kb * 8 + nt) << 9));
            acc[nt] = __builtin_amdgcn_wmma_f32_16x16x32_bf16(false, a1[kb], false, b,
                                                              (short)0, acc[nt], false, false);
        }
    }

    // ---- ReLU -> LDS (per-wave 16x128 bf16 row-major tile).
    //      C layout: acc[nt][r] = C[row 8*hi + r][col 16*nt + m].
    __bf16* hl = &lds_h[wave * 16 * DHID];
#pragma unroll
    for (int nt = 0; nt < 8; ++nt) {
#pragma unroll
        for (int r = 0; r < 8; ++r) {
            float v = acc[nt][r];
            v = v > 0.0f ? v : 0.0f;
            hl[(hi * 8 + r) * DHID + nt * 16 + m] = (__bf16)v;
        }
    }

    __syncthreads();   // order DS writes before cross-lane DS reads

    // ---- Re-load intermediate tile from LDS in A layout.
    v16bf a2[4];
#pragma unroll
    for (int kb = 0; kb < 4; ++kb) {
        const v8bf lo = *(const v8bf*)(hl + m * DHID + kb * 32 + 8 * hi);
        const v8bf hc = *(const v8bf*)(hl + m * DHID + kb * 32 + 16 + 8 * hi);
        v16bf a;
#pragma unroll
        for (int i = 0; i < 8; ++i) { a[i] = lo[i]; a[8 + i] = hc[i]; }
        a2[kb] = a;
    }

    // ---- GEMM2: bias-seeded accumulators, kb-outer for ILP.
#pragma unroll
    for (int nt = 0; nt < 8; ++nt) {
        const float bb = b2[nt * 16 + m];
        acc[nt] = (v8f){bb, bb, bb, bb, bb, bb, bb, bb};
    }
#pragma unroll
    for (int kb = 0; kb < 4; ++kb) {
#pragma unroll
        for (int nt = 0; nt < 8; ++nt) {
            const v16bf b = *(const v16bf*)(w2p + ((kb * 8 + nt) << 9));
            acc[nt] = __builtin_amdgcn_wmma_f32_16x16x32_bf16(false, a2[kb], false, b,
                                                              (short)0, acc[nt], false, false);
        }
    }

    // ---- Store f32 result (row-guarded; EXEC stays all-1s for WMMA).
#pragma unroll
    for (int nt = 0; nt < 8; ++nt) {
#pragma unroll
        for (int r = 0; r < 8; ++r) {
            const long orow = rowBase + hi * 8 + r;
            if (orow < nrows)
                out[orow * DHID + nt * 16 + m] = acc[nt][r];
        }
    }
}

// ---------------------------------------------------------------------------
extern "C" void kernel_launch(void* const* d_in, const int* in_sizes, int n_in,
                              void* d_out, int out_size, void* d_ws, size_t ws_size,
                              hipStream_t stream)
{
    const float* x  = (const float*)d_in[0];
    const int*   ei = (const int*)d_in[1];
    const float* W1 = (const float*)d_in[2];
    const float* b1 = (const float*)d_in[3];
    const float* W2 = (const float*)d_in[4];
    const float* b2 = (const float*)d_in[5];
    float* out = (float*)d_out;

    const int N = in_sizes[0] / DHID;            // 50000
    const int E = in_sizes[1] / 2;               // 800000
    const int L = in_sizes[2] / (DHID * DHID);   // 3

    const size_t nd_bytes = (size_t)N * DHID * sizeof(float);
    float*  bufA = (float*)d_ws;                              // hnext
    float*  bufB = (float*)((char*)d_ws + nd_bytes);          // agg/hsum
    __bf16* wsw  = (__bf16*)((char*)d_ws + 2 * nd_bytes);     // swizzled weights

    // One-time (per call) weight convert+swizzle: L*2*128*128 bf16 elements.
    const int wtotal = L * 2 * DHID * DHID;
    gin_wprep<<<(wtotal + 255) / 256, 256, 0, stream>>>(W1, W2, wsw, wtotal);

    const int mat_elems = 4 * 8 * 512;  // 16384 bf16 per 128x128 matrix

    const float* hcur = x;
    for (int l = 0; l < L; ++l) {
        // agg = h  (GIN eps=0: h_i + sum of neighbors)
        hipMemcpyAsync(bufB, hcur, nd_bytes, hipMemcpyDeviceToDevice, stream);
        // agg[dst] += h[src]
        const long sthreads = (long)E * 32;
        gin_scatter<<<(int)((sthreads + 255) / 256), 256, 0, stream>>>(
            hcur, ei, ei + E, bufB, E);
        // hnext = relu(agg*W1 + b1)*W2 + b2
        float* hout = (l == L - 1) ? out : bufA;
        gin_mlp<<<(N + 127) / 128, 256, 0, stream>>>(
            bufB,
            wsw + (size_t)(l * 2 + 0) * mat_elems, b1 + (size_t)l * DHID,
            wsw + (size_t)(l * 2 + 1) * mat_elems, b2 + (size_t)l * DHID,
            hout, N);
        hcur = bufA;
    }
}
